// STGCN_85856396247963
// MI455X (gfx1250) — matrix operand order repaired and checked by reading, not compiled, not run on hardware
//
#include <hip/hip_runtime.h>
#include <hip/hip_bf16.h>

#define TT 12
#define NN 10000
#define FF 256
#define EE 160000
#define TPAD 14              // TT + 2 causal pad slices
static constexpr size_t NF = (size_t)NN * FF;   // 2,560,000

#define AROW 776             // 768 + 8 ushort pad -> bank-conflict-free ds_load_b128
#define MT 4                 // row tiles (of 16) per block; B L2 traffic / MT

typedef __attribute__((ext_vector_type(16))) __bf16 bf16x16;
typedef __attribute__((ext_vector_type(8)))  float  f32x8;

union Frag16 { bf16x16 v; uint4 q[2]; };

__device__ __forceinline__ unsigned short f32_to_bf16(float f) {
    unsigned int u = __float_as_uint(f);
    u += 0x7fffu + ((u >> 16) & 1u);   // round-to-nearest-even
    return (unsigned short)(u >> 16);
}
__device__ __forceinline__ float bf16_to_f32(unsigned short h) {
    return __uint_as_float(((unsigned int)h) << 16);
}

// ---------------------------------------------------------------------------
// Weight prep: conv_w/gate_w [L][O=256][C=256][K=3] f32 ->
// Wt[L][o'=512][ k_time*256 + c ] bf16  (o'<256 conv, o'>=256 gate)
// ---------------------------------------------------------------------------
__global__ __launch_bounds__(256) void prep_weights(
    const float* __restrict__ convW, const float* __restrict__ gateW,
    unsigned short* __restrict__ wts)
{
    int tid = blockIdx.x * 256 + threadIdx.x;          // 3*512*768 threads
    int i  = tid / (512 * 768);
    int r  = tid - i * 512 * 768;
    int o  = r / 768;
    int kc = r - o * 768;
    int kt = kc >> 8;
    int c  = kc & 255;
    const float* W = (o < 256) ? convW : gateW;
    int oo = o & 255;
    float f = W[(((size_t)i * 256 + oo) * 256 + c) * 3 + kt];
    wts[tid] = f32_to_bf16(f);
}

// ---------------------------------------------------------------------------
// CSR build for pull-based segment_sum:  off[t][n+1] = in-degree histogram
// ---------------------------------------------------------------------------
__global__ __launch_bounds__(256) void hist_kernel(
    const int* __restrict__ adj, int* __restrict__ off)
{
    int tid = blockIdx.x * 256 + threadIdx.x;          // T*E threads
    int e = tid % EE;
    int t = tid / EE;
    int dst = adj[(size_t)t * 2 * EE + EE + e];
    atomicAdd(&off[(size_t)t * (NN + 1) + dst + 1], 1);
}

// In-place inclusive scan of off[t][1..NN]; one block per timestep.
__global__ __launch_bounds__(1024) void scan_offsets(int* __restrict__ off)
{
    __shared__ int part[1024];
    int* a = off + (size_t)blockIdx.x * (NN + 1);
    const int tid = threadIdx.x;
    int loc[10];
    int s = 0;
    if (tid < 1000) {
        int base = 1 + tid * 10;
        #pragma unroll
        for (int i = 0; i < 10; ++i) { s += a[base + i]; loc[i] = s; }
    }
    part[tid] = (tid < 1000) ? s : 0;
    __syncthreads();
    for (int d = 1; d < 1024; d <<= 1) {         // Hillis-Steele inclusive scan
        int v = (tid >= d) ? part[tid - d] : 0;
        __syncthreads();
        part[tid] += v;
        __syncthreads();
    }
    if (tid < 1000) {
        int pre = part[tid] - s;                 // exclusive prefix of this chunk
        int base = 1 + tid * 10;
        #pragma unroll
        for (int i = 0; i < 10; ++i) a[base + i] = pre + loc[i];
    }
    if (tid == 0) a[0] = 0;
}

__global__ __launch_bounds__(256) void fill_kernel(
    const int* __restrict__ adj, const int* __restrict__ off,
    int* __restrict__ cursor, int* __restrict__ srcids)
{
    int tid = blockIdx.x * 256 + threadIdx.x;          // T*E threads
    int e = tid % EE;
    int t = tid / EE;
    const int* at = adj + (size_t)t * 2 * EE;
    int src = at[e];
    int dst = at[EE + e];
    int pos = atomicAdd(&cursor[t * NN + dst], 1);
    srcids[(size_t)t * EE + off[(size_t)t * (NN + 1) + dst] + pos] = src;
}

// Sort each node's (avg ~16 entry) src list -> deterministic summation order.
__global__ __launch_bounds__(256) void sort_lists(
    const int* __restrict__ off, int* __restrict__ srcids)
{
    int tid = blockIdx.x * 256 + threadIdx.x;          // T*N threads
    if (tid >= TT * NN) return;
    int n = tid % NN;
    int t = tid / NN;
    int b  = off[(size_t)t * (NN + 1) + n];
    int e2 = off[(size_t)t * (NN + 1) + n + 1];
    int* s = srcids + (size_t)t * EE;
    for (int i = b + 1; i < e2; ++i) {
        int key = s[i];
        int j = i - 1;
        while (j >= b && s[j] > key) { s[j + 1] = s[j]; --j; }
        s[j + 1] = key;
    }
}

// ---------------------------------------------------------------------------
// Pull-based A @ X per timestep: y[t][n] = sum_{e: dst==n} x[t][src(e)]
// thread = (t, n, 4-channel chunk); no float atomics, fully coalesced.
// ---------------------------------------------------------------------------
__global__ __launch_bounds__(256) void gather_ax(
    const float* __restrict__ x, const int* __restrict__ off,
    const int* __restrict__ srcids, float* __restrict__ y)
{
    long long tid = (long long)blockIdx.x * 256 + threadIdx.x;   // T*N*64
    int c4 = (int)(tid & 63);
    long long tn = tid >> 6;
    int n = (int)(tn % NN);
    int t = (int)(tn / NN);
    int b  = off[(size_t)t * (NN + 1) + n];
    int e2 = off[(size_t)t * (NN + 1) + n + 1];
    const int* s = srcids + (size_t)t * EE;
    const float* xt = x + (size_t)t * NF + (size_t)c4 * 4;
    float4 acc = {0.f, 0.f, 0.f, 0.f};
    for (int i = b; i < e2; ++i) {
        const float4 v = *(const float4*)(xt + (size_t)s[i] * FF);
        acc.x += v.x; acc.y += v.y; acc.z += v.z; acc.w += v.w;
    }
    *(float4*)(y + (size_t)t * NF + (size_t)n * FF + c4 * 4) = acc;
}

// ---------------------------------------------------------------------------
// f32 [T,N,F] -> bf16 [(T+2),N,F] with two leading zero slices (causal pad)
// ---------------------------------------------------------------------------
__global__ __launch_bounds__(256) void cvt_pad(
    const float* __restrict__ in, unsigned short* __restrict__ out)
{
    size_t tid = (size_t)blockIdx.x * 256 + threadIdx.x;   // (T+2)*NF/4 threads
    size_t idx = tid * 4;
    size_t s = idx / NF;
    ushort4 o;
    if (s < 2) {
        o.x = o.y = o.z = o.w = 0;
    } else {
        float4 f = *(const float4*)(in + (idx - 2 * NF));
        o.x = f32_to_bf16(f.x); o.y = f32_to_bf16(f.y);
        o.z = f32_to_bf16(f.z); o.w = f32_to_bf16(f.w);
    }
    *(ushort4*)(out + idx) = o;
}

// ---------------------------------------------------------------------------
// Gated TCN layer as WMMA GEMM with M-blocking.
//   Block = 512 threads = 16 waves; block owns MT=4 row tiles (64 rows).
//   All 4 A panels (16x768 bf16 each, row stride 776) staged in dynamic LDS
//   (~97 KB). Each wave owns one conv/gate col-tile pair and accumulates 4
//   row tiles, so every B fragment fetched from L2 feeds 4 WMMAs. 24 K-steps
//   x 8 v_wmma_f32_16x16x32_bf16 per wave; GLU fused, bf16 out.
// ---------------------------------------------------------------------------
__global__ __launch_bounds__(512) void tcn_gemm(
    const unsigned short* __restrict__ embIn,   // [(T+2),N,F] bf16
    unsigned short* __restrict__ embOut,        // [(T+2),N,F] bf16
    const unsigned short* __restrict__ wts,     // [512][768] bf16 (this layer)
    const float* __restrict__ convB,            // [F]
    const float* __restrict__ gateB)            // [F]
{
    extern __shared__ __align__(16) unsigned short As[];   // MT*16*AROW ushorts

    const int lane = threadIdx.x & 31;
    const int wave = threadIdx.x >> 5;          // 0..15 -> col-tile pair

    // ---- cooperative stage of MT A panels:
    // As[m][r][kt*256+c] = embIn[t_m+kt][n0_m+r][c]
    #pragma unroll
    for (int it = 0; it < MT * 3; ++it) {       // MT*16*96 uint4 chunks / 512 thr
        int q   = it * 512 + threadIdx.x;
        int m   = q / 1536;
        int rem = q - m * 1536;
        int r   = rem / 96;
        int cc  = rem - r * 96;
        int kt  = cc >> 5;
        int c   = (cc & 31) << 3;
        int g   = blockIdx.x * MT + m;          // global row-tile id (0..7499)
        int tm  = g % TT;
        int n0m = (g / TT) << 4;
        const unsigned short* src = embIn + (size_t)(tm + kt) * NF
                                  + (size_t)(n0m + r) * FF + c;
        *(uint4*)&As[(m * 16 + r) * AROW + (kt << 8) + c] = *(const uint4*)src;
    }
    __syncthreads();

    const int half = (lane < 16) ? 0 : 1;
    const int colC = (wave << 4) + (lane & 15); // conv column; gate = +256

    f32x8 accC[MT];
    f32x8 accG[MT];
    #pragma unroll
    for (int m = 0; m < MT; ++m) { accC[m] = (f32x8){}; accG[m] = (f32x8){}; }

    const unsigned short* wC = wts + (size_t)colC * 768 + half * 16;
    const unsigned short* wG = wts + (size_t)(colC + 256) * 768 + half * 16;
    const unsigned short* aBase = As + (lane & 15) * AROW + half * 16;

    for (int kk = 0; kk < 24; ++kk) {
        Frag16 bc, bg;
        const unsigned short* pc = wC + kk * 32;
        bc.q[0] = *(const uint4*)(pc);
        bc.q[1] = *(const uint4*)(pc + 8);
        const unsigned short* pg = wG + kk * 32;
        bg.q[0] = *(const uint4*)(pg);
        bg.q[1] = *(const uint4*)(pg + 8);
        #pragma unroll
        for (int m = 0; m < MT; ++m) {
            Frag16 a;
            const unsigned short* pa = aBase + m * 16 * AROW + kk * 32;
            a.q[0] = *(const uint4*)(pa);          // K = base .. base+7
            a.q[1] = *(const uint4*)(pa + 8);      // K = base+8 .. base+15
            accC[m] = __builtin_amdgcn_wmma_f32_16x16x32_bf16(
                          false, a.v, false, bc.v, (short)0, accC[m], false, false);
            accG[m] = __builtin_amdgcn_wmma_f32_16x16x32_bf16(
                          false, a.v, false, bg.v, (short)0, accG[m], false, false);
        }
    }

    const float bcv = convB[colC];
    const float bgv = gateB[colC];
    #pragma unroll
    for (int m = 0; m < MT; ++m) {
        int g   = blockIdx.x * MT + m;
        int tm  = g % TT;
        int n0m = (g / TT) << 4;
        unsigned short* outSlice = embOut + (size_t)(tm + 2) * NF;
        #pragma unroll
        for (int v = 0; v < 8; ++v) {
            int row = n0m + v + half * 8;
            float c = accC[m][v] + bcv;
            float gg = accG[m][v] + bgv;
            float o = c * (1.0f / (1.0f + __expf(-gg)));   // GLU
            outSlice[(size_t)row * FF + colC] = f32_to_bf16(o);
        }
    }
}

// ---------------------------------------------------------------------------
// Head: out[n] = log_softmax( emb_last[n] @ lin_w.T + lin_b )  (64 threads/node)
// ---------------------------------------------------------------------------
__global__ __launch_bounds__(64) void head_kernel(
    const unsigned short* __restrict__ emb,   // [(T+2),N,F] bf16; last slice = 13
    const float* __restrict__ lw,             // [64][256]
    const float* __restrict__ lb,             // [64]
    float* __restrict__ out)                  // [N][64]
{
    __shared__ float xr[FF];
    __shared__ float red[64];
    int n = blockIdx.x;
    int o = threadIdx.x;

    const unsigned short* px = emb + (size_t)(TPAD - 1) * NF + (size_t)n * FF;
    ushort4 h = *(const ushort4*)(px + o * 4);
    xr[o * 4 + 0] = bf16_to_f32(h.x);
    xr[o * 4 + 1] = bf16_to_f32(h.y);
    xr[o * 4 + 2] = bf16_to_f32(h.z);
    xr[o * 4 + 3] = bf16_to_f32(h.w);
    __syncthreads();

    float acc = lb[o];
    const float* w = lw + o * FF;
    #pragma unroll 8
    for (int c = 0; c < FF; ++c) acc += xr[c] * w[c];

    red[o] = acc;
    __syncthreads();
    for (int s = 32; s > 0; s >>= 1) {
        if (o < s) red[o] = fmaxf(red[o], red[o + s]);
        __syncthreads();
    }
    float m = red[0];
    __syncthreads();
    red[o] = __expf(acc - m);
    __syncthreads();
    for (int s = 32; s > 0; s >>= 1) {
        if (o < s) red[o] += red[o + s];
        __syncthreads();
    }
    out[(size_t)n * 64 + o] = acc - m - logf(red[0]);
}

// ---------------------------------------------------------------------------
extern "C" void kernel_launch(void* const* d_in, const int* in_sizes, int n_in,
                              void* d_out, int out_size, void* d_ws, size_t ws_size,
                              hipStream_t stream) {
    const float* feats = (const float*)d_in[0];   // [T,N,F]
    const int*   adjs  = (const int*)d_in[1];     // [T,2,E]
    const float* convW = (const float*)d_in[2];   // [3,256,256,3]
    const float* convB = (const float*)d_in[3];   // [3,256]
    const float* gateW = (const float*)d_in[4];
    const float* gateB = (const float*)d_in[5];
    const float* linW  = (const float*)d_in[6];   // [64,256]
    const float* linB  = (const float*)d_in[7];   // [64]
    float* out = (float*)d_out;

    // Workspace layout (~257 MB):
    //   region0: buf1 f32 [T,N,F]  ->  later embA bf16 [(T+2),N,F]
    //   region1: buf2 f32 [T,N,F]  ->  later embB bf16 [(T+2),N,F]
    //   region2: weights bf16 [3][512][768]
    //   region3: CSR off[T][N+1], cursor[T][N], srcids[T][E]  (ints)
    float* buf1 = (float*)d_ws;
    float* buf2 = buf1 + (size_t)TT * NF;
    unsigned short* embA = (unsigned short*)buf1;
    unsigned short* embB = (unsigned short*)buf2;
    unsigned short* wts  = (unsigned short*)(buf2 + (size_t)TT * NF);
    int* off    = (int*)(wts + (size_t)3 * 512 * 768);
    int* cursor = off + (size_t)TT * (NN + 1);
    int* srcids = cursor + (size_t)TT * NN;

    prep_weights<<<4608, 256, 0, stream>>>(convW, gateW, wts);

    // ---- build per-timestep inverse CSR (deterministic via per-list sort)
    hipMemsetAsync(off, 0, (size_t)TT * (NN + 1) * sizeof(int), stream);
    hist_kernel<<<(TT * EE) / 256, 256, 0, stream>>>(adjs, off);
    scan_offsets<<<TT, 1024, 0, stream>>>(off);
    hipMemsetAsync(cursor, 0, (size_t)TT * NN * sizeof(int), stream);
    fill_kernel<<<(TT * EE) / 256, 256, 0, stream>>>(adjs, off, cursor, srcids);
    sort_lists<<<(TT * NN + 255) / 256, 256, 0, stream>>>(off, srcids);

    // ---- two pull-based propagation layers (no float atomics, no big memsets)
    gather_ax<<<(int)((size_t)TT * NN * 64 / 256), 256, 0, stream>>>(feats, off, srcids, buf1);
    gather_ax<<<(int)((size_t)TT * NN * 64 / 256), 256, 0, stream>>>(buf1, off, srcids, buf2);

    // bf16 + causal pad into embA (region0; buf1 dead now)
    cvt_pad<<<(int)((TPAD * NF) / 4 / 256), 256, 0, stream>>>(buf2, embA);
    // zero embB pad slices (region1; buf2 dead after cvt_pad)
    hipMemsetAsync(embB, 0, 2 * NF * sizeof(unsigned short), stream);

    const size_t ldsBytes = (size_t)MT * 16 * AROW * sizeof(unsigned short); // 99,328 B
    // Three gated TCN layers (ping-pong embA <-> embB); pads stay zero.
    tcn_gemm<<<1875, 512, ldsBytes, stream>>>(embA, embB, wts + 0 * 512 * 768,
                                              convB + 0 * FF, gateB + 0 * FF);
    tcn_gemm<<<1875, 512, ldsBytes, stream>>>(embB, embA, wts + 1 * 512 * 768,
                                              convB + 1 * FF, gateB + 1 * FF);
    tcn_gemm<<<1875, 512, ldsBytes, stream>>>(embA, embB, wts + 2 * 512 * 768,
                                              convB + 2 * FF, gateB + 2 * FF);

    head_kernel<<<NN, 64, 0, stream>>>(embB, linW, linB, out);
}
